// BatchedGAT_71571335020986
// MI455X (gfx1250) — matrix operand order repaired and checked by reading, not compiled, not run on hardware
//
#include <hip/hip_runtime.h>

typedef __attribute__((ext_vector_type(16))) _Float16 v16h;
typedef __attribute__((ext_vector_type(8)))  _Float16 v8h;
typedef __attribute__((ext_vector_type(8)))  float    v8f;
typedef __attribute__((ext_vector_type(4)))  float    f32x4;

#define BQ 16
#define NQ 1024
#define DQ 256
#define HQ 4
#define FQ 64
#define HF 256
#define XN ((size_t)BQ * NQ * DQ)   // 4,194,304 elements of x / h

// wave32 lane broadcast / exchange via ds_bpermute (index in bytes)
__device__ __forceinline__ float lane_f(float v, int srcLane) {
  return __int_as_float(__builtin_amdgcn_ds_bpermute(srcLane << 2, __float_as_int(v)));
}

// -----------------------------------------------------------------------------
// K0: one-shot f16 preconversion.
//   xh[i]       = (f16) x[i]                    (row-major, 8 MiB)
//   WT[n][k]    = (f16) W[k][n]                 (transposed, 128 KiB)
// -----------------------------------------------------------------------------
__global__ __launch_bounds__(256) void gat_cvt(
    const float* __restrict__ x, const float* __restrict__ W,
    _Float16* __restrict__ xh, _Float16* __restrict__ WT) {
  const size_t tid = (size_t)blockIdx.x * 256 + threadIdx.x;
  if (tid < XN) {
    xh[tid] = (_Float16)x[tid];
  } else {
    const int idx = (int)(tid - XN);          // 0..65535
    const int n = idx >> 8, k = idx & 255;
    WT[(size_t)n * DQ + k] = (_Float16)W[(size_t)k * HF + n];
  }
}

// -----------------------------------------------------------------------------
// K1: h[b*N+n, 256] = x @ W.  One 16x16 tile per wave, 8 x wmma_f32_16x16x32_f16.
// A: two 16B f16 loads/chunk.  B: two 16B f16 loads/chunk (from WT).
// -----------------------------------------------------------------------------
__global__ __launch_bounds__(128) void gat_gemm_h(
    const _Float16* __restrict__ xh, const _Float16* __restrict__ WT,
    float* __restrict__ h) {
  const int tid  = threadIdx.x;
  const int lane = tid & 31;
  const int lm   = lane & 15;
  const int hl   = lane >> 4;
  const int w    = blockIdx.x * 4 + (tid >> 5);   // 0..16383 tiles
  const int tileM = w >> 4;                       // 16-row tiles of B*N
  const int tileN = w & 15;                       // 16-col tiles of 256

  const _Float16* xrow = xh + (size_t)(tileM * 16 + lm) * DQ;
  const int col = tileN * 16 + lm;
  const _Float16* wcol = WT + (size_t)col * DQ;

  v8f c = {};
#pragma unroll
  for (int k0 = 0; k0 < DQ; k0 += 32) {
    // A fragment (16x32): half 0 -> K 0..7 & 16..23, half 1 -> K 8..15 & 24..31
    v8h lo = *(const v8h*)(xrow + k0 + hl * 8);
    v8h hi = *(const v8h*)(xrow + k0 + 16 + hl * 8);
    v16h av;
#pragma unroll
    for (int i = 0; i < 8; ++i) { av[i] = lo[i]; av[8 + i] = hi[i]; }
    // B fragment (32x16): lane column = col, same K order as A
    v8h blo = *(const v8h*)(wcol + k0 + hl * 8);
    v8h bhi = *(const v8h*)(wcol + k0 + 16 + hl * 8);
    v16h bv;
#pragma unroll
    for (int i = 0; i < 8; ++i) { bv[i] = blo[i]; bv[8 + i] = bhi[i]; }
    c = __builtin_amdgcn_wmma_f32_16x16x32_f16(false, av, false, bv, (short)0, c,
                                               false, false);
  }
#pragma unroll
  for (int v = 0; v < 8; ++v)   // C layout: row = v + 8*half, col = lm
    h[(size_t)(tileM * 16 + v + hl * 8) * HF + col] = c[v];
}

// -----------------------------------------------------------------------------
// K2: e_src/e_dst in head-major [B,H,N] (f32) and hT[b,h,f,n] (f16 transpose)
// -----------------------------------------------------------------------------
__global__ __launch_bounds__(256) void gat_prep(
    const float* __restrict__ h, const float* __restrict__ a_src,
    const float* __restrict__ a_dst, float* __restrict__ es,
    float* __restrict__ ed, _Float16* __restrict__ hT) {
  const int gid = blockIdx.x * 256 + threadIdx.x;   // b*1024 + n
  const int b = gid >> 10;
  const int n = gid & 1023;
  const float* hr = h + (size_t)gid * HF;
#pragma unroll
  for (int hh = 0; hh < HQ; ++hh) {
    float s1 = 0.f, s2 = 0.f;
#pragma unroll 8
    for (int f = 0; f < FQ; ++f) {
      float v = hr[hh * FQ + f];
      s1 += v * a_src[hh * FQ + f];
      s2 += v * a_dst[hh * FQ + f];
    }
    es[((b * HQ + hh) << 10) + n] = s1;
    ed[((b * HQ + hh) << 10) + n] = s2;
  }
  for (int c = 0; c < HF; ++c)   // coalesced across n for each (h,f)
    hT[((size_t)(b * HQ + (c >> 6)) * FQ + (c & 63)) * NQ + n] = (_Float16)hr[c];
}

// -----------------------------------------------------------------------------
// K3: fused masked-softmax + aggregation (flash-style, adj streamed once, NT).
// One wave = (batch b, 16-row i-tile, head hh). j streamed in chunks of 32.
// -----------------------------------------------------------------------------
__global__ __launch_bounds__(128) void gat_attn(
    const float* __restrict__ adj, const float* __restrict__ es,
    const float* __restrict__ edst, const _Float16* __restrict__ hT,
    const float* __restrict__ bias, float* __restrict__ out) {
  const int tid  = threadIdx.x;
  const int lane = tid & 31;
  const int lm   = lane & 15;
  const int hl   = lane >> 4;
  const int w    = blockIdx.x * 4 + (tid >> 5);   // 0..4095
  const int b  = w >> 8;
  const int it = (w >> 2) & 63;
  const int hh = w & 3;

  const int i0  = it * 16;
  const int row = i0 + lm;
  const float* arow = adj + ((size_t)(b << 10) + row) * NQ;
  const float* esr  = es + ((b * HQ + hh) << 10);
  const float  edv  = edst[((b * HQ + hh) << 10) + row];
  const _Float16* hTh = hT + (size_t)(b * HQ + hh) * FQ * NQ;

  float m = -__builtin_inff();
  float lsum = 0.f;
  v8f acc0 = {}, acc1 = {}, acc2 = {}, acc3 = {};

  for (int jb = 0; jb < NQ; jb += 32) {
    f32x4 e0 = *(const f32x4*)(esr + jb + hl * 8);
    f32x4 e1 = *(const f32x4*)(esr + jb + hl * 8 + 4);
    f32x4 e2 = *(const f32x4*)(esr + jb + 16 + hl * 8);
    f32x4 e3 = *(const f32x4*)(esr + jb + 16 + hl * 8 + 4);
    // adj is streamed exactly once -> non-temporal (keep L2 for hT/es reuse)
    f32x4 g0 = __builtin_nontemporal_load((const f32x4*)(arow + jb + hl * 8));
    f32x4 g1 = __builtin_nontemporal_load((const f32x4*)(arow + jb + hl * 8 + 4));
    f32x4 g2 = __builtin_nontemporal_load((const f32x4*)(arow + jb + 16 + hl * 8));
    f32x4 g3 = __builtin_nontemporal_load((const f32x4*)(arow + jb + 16 + hl * 8 + 4));

    // LeakyReLU logits + adjacency mask (finite -1e9 matches reference)
    float l[16];
#pragma unroll
    for (int i = 0; i < 4; ++i) {
      float t;
      t = edv + e0[i]; t = (t >= 0.f) ? t : 0.2f * t; l[i]      = (g0[i] > 0.5f) ? t : -1e9f;
      t = edv + e1[i]; t = (t >= 0.f) ? t : 0.2f * t; l[4 + i]  = (g1[i] > 0.5f) ? t : -1e9f;
      t = edv + e2[i]; t = (t >= 0.f) ? t : 0.2f * t; l[8 + i]  = (g2[i] > 0.5f) ? t : -1e9f;
      t = edv + e3[i]; t = (t >= 0.f) ? t : 0.2f * t; l[12 + i] = (g3[i] > 0.5f) ? t : -1e9f;
    }

    // online softmax: row max across both lane-halves
    float cm = l[0];
#pragma unroll
    for (int i = 1; i < 16; ++i) cm = fmaxf(cm, l[i]);
    cm = fmaxf(cm, lane_f(cm, lane ^ 16));
    float mn = fmaxf(m, cm);
    float scale = __expf(m - mn);   // exp(-inf)=0 on first chunk
    m = mn;

    // probabilities -> A fragment (element order matches 16-bit A layout)
    float ps = 0.f;
    v16h av;
#pragma unroll
    for (int i = 0; i < 16; ++i) {
      float p = __expf(l[i] - mn);
      ps += p;
      av[i] = (_Float16)p;
    }
    ps += lane_f(ps, lane ^ 16);
    lsum = lsum * scale + ps;

    // rescale accumulators with per-row factor (row stats live in lanes 0..15)
#pragma unroll
    for (int v = 0; v < 8; ++v) {
      float f = lane_f(scale, v + hl * 8);
      acc0[v] *= f; acc1[v] *= f; acc2[v] *= f; acc3[v] *= f;
    }

    // B fragments: hT[b,h,f, jb + hl*8 .. +7] and [jb + 16 + hl*8 .. +7] (same K order as A)
    const _Float16* bbase = hTh + jb + hl * 8;
    v16h b0, b1, b2, b3;
#pragma unroll
    for (int i = 0; i < 8; ++i) {
      b0[i] = bbase[(size_t)(0 * 16 + lm) * NQ + i]; b0[8 + i] = bbase[(size_t)(0 * 16 + lm) * NQ + 16 + i];
      b1[i] = bbase[(size_t)(1 * 16 + lm) * NQ + i]; b1[8 + i] = bbase[(size_t)(1 * 16 + lm) * NQ + 16 + i];
      b2[i] = bbase[(size_t)(2 * 16 + lm) * NQ + i]; b2[8 + i] = bbase[(size_t)(2 * 16 + lm) * NQ + 16 + i];
      b3[i] = bbase[(size_t)(3 * 16 + lm) * NQ + i]; b3[8 + i] = bbase[(size_t)(3 * 16 + lm) * NQ + 16 + i];
    }
    acc0 = __builtin_amdgcn_wmma_f32_16x16x32_f16(false, av, false, b0, (short)0, acc0, false, false);
    acc1 = __builtin_amdgcn_wmma_f32_16x16x32_f16(false, av, false, b1, (short)0, acc1, false, false);
    acc2 = __builtin_amdgcn_wmma_f32_16x16x32_f16(false, av, false, b2, (short)0, acc2, false, false);
    acc3 = __builtin_amdgcn_wmma_f32_16x16x32_f16(false, av, false, b3, (short)0, acc3, false, false);
  }

  float inv = 1.0f / lsum;
#pragma unroll
  for (int v = 0; v < 8; ++v) {
    float iv = lane_f(inv, v + hl * 8);
    size_t orow = ((size_t)(b << 10) + i0 + v + hl * 8) * HF + hh * FQ;
    out[orow + 0 * 16 + lm] = acc0[v] * iv + bias[hh * FQ + 0 * 16 + lm];
    out[orow + 1 * 16 + lm] = acc1[v] * iv + bias[hh * FQ + 1 * 16 + lm];
    out[orow + 2 * 16 + lm] = acc2[v] * iv + bias[hh * FQ + 2 * 16 + lm];
    out[orow + 3 * 16 + lm] = acc3[v] * iv + bias[hh * FQ + 3 * 16 + lm];
  }
}

extern "C" void kernel_launch(void* const* d_in, const int* in_sizes, int n_in,
                              void* d_out, int out_size, void* d_ws, size_t ws_size,
                              hipStream_t stream) {
  const float* x     = (const float*)d_in[0];
  const float* adj   = (const float*)d_in[1];
  const float* W     = (const float*)d_in[2];
  const float* a_src = (const float*)d_in[3];
  const float* a_dst = (const float*)d_in[4];
  const float* bias  = (const float*)d_in[5];
  float* out = (float*)d_out;

  // workspace: h (16 MiB f32) | es | ed (256 KiB) | hT (8 MiB f16)
  //            | xh (8 MiB f16) | WT (128 KiB f16)
  float* h  = (float*)d_ws;                              // 16384*256 f32
  float* es = h + XN;                                    // B*H*N f32
  float* ed = es + (size_t)BQ * HQ * NQ;                 // B*H*N f32
  _Float16* hT = (_Float16*)(ed + (size_t)BQ * HQ * NQ); // B*H*F*N f16
  _Float16* xh = hT + XN;                                // B*N*D f16
  _Float16* WT = xh + XN;                                // 256*256 f16

  const int cvt_elems = (int)(XN + 65536);
  gat_cvt<<<cvt_elems / 256, 256, 0, stream>>>(x, W, xh, WT);
  gat_gemm_h<<<4096, 128, 0, stream>>>(xh, WT, h);
  gat_prep<<<64, 256, 0, stream>>>(h, a_src, a_dst, es, ed, hT);
  gat_attn<<<1024, 128, 0, stream>>>(adj, es, ed, hT, bias, out);
}
